// DGN3_70428873720437
// MI455X (gfx1250) — matrix operand order repaired and checked by reading, not compiled, not run on hardware
//
#include <hip/hip_runtime.h>
#include <hip/hip_bf16.h>
#include <math.h>

#define B_  4
#define T_  2048
#define D_  256
#define K_  8
#define R_  3
#define NEGV (-1e9f)
#define TTILES (T_/16)   // 128 query tiles per batch

typedef __attribute__((ext_vector_type(16))) __bf16        v16bf;
typedef __attribute__((ext_vector_type(8)))  float          v8f;
typedef __attribute__((ext_vector_type(8)))  unsigned short u16x8;
typedef __attribute__((ext_vector_type(16))) unsigned short u16x16;
typedef __attribute__((ext_vector_type(4)))  unsigned int   u32x4;
typedef __attribute__((ext_vector_type(4)))  int            i32x4;
typedef __attribute__((ext_vector_type(8)))  int            i32x8;

struct APairBits { u16x8 lo; u16x8 hi; };   // 32 bytes == sizeof(v16bf)

__device__ __forceinline__ unsigned short f2bf(float f) {
    unsigned u = __float_as_uint(f);
    u += 0x7FFFu + ((u >> 16) & 1u);        // round-to-nearest-even
    return (unsigned short)(u >> 16);
}
__device__ __forceinline__ float sigm(float v) { return 1.0f / (1.0f + expf(-v)); }

// ---------------------------------------------------------------------------
// TDM: DMA one 16-row x 512-byte key tile (bf16) from global into LDS.
// D# group0: count=1, lds_addr, global_addr, type=2 ("image").
// D# group1: data_size=4B units; tile 128x16, tensor_dim0(_stride)=128,
//            tensor_dim1=16 (tile fully in-bounds), no padding/iterate/mask.
// ---------------------------------------------------------------------------
__device__ __forceinline__ void tdm_load_tile(const unsigned short* gptr,
                                              unsigned ldsoff) {
    unsigned long long ga = (unsigned long long)(size_t)gptr;
    u32x4 g0;
    g0.x = 1u;                                        // count=1 (valid descriptor)
    g0.y = ldsoff;                                    // LDS byte address
    g0.z = (unsigned)ga;                              // global_addr[31:0]
    g0.w = (unsigned)(ga >> 32) | 0x80000000u;        // global_addr[56:32] | type=2
    i32x8 g1 = { 0x00020000,        // data_size=4B
                 0x00800000,        // tensor_dim0 = 128 (low16 << 16)
                 0x00100000,        // tensor_dim1 = 16  (low16 << 16)
                 0x00800000,        // tile_dim0   = 128 (<< 16)
                 16,                // tile_dim1   = 16
                 128,               // tensor_dim0_stride = 128 dwords
                 0, 0 };
    i32x4 z4 = {0, 0, 0, 0};
#if __clang_major__ >= 23
    i32x8 z8 = {0, 0, 0, 0, 0, 0, 0, 0};
    __builtin_amdgcn_tensor_load_to_lds(g0, g1, z4, z4, z8, 0);
#else
    __builtin_amdgcn_tensor_load_to_lds(g0, g1, z4, z4, 0);
#endif
}

// ---------------------------------------------------------------------------
// f32 -> bf16 bit conversion of the initial h (= x)
// ---------------------------------------------------------------------------
__global__ void DGN3_cvt_bf16(const float* __restrict__ src,
                              unsigned short* __restrict__ dst, int n) {
    int i = blockIdx.x * blockDim.x + threadIdx.x;
    if (i < n) dst[i] = f2bf(src[i]);
}

// ---------------------------------------------------------------------------
// One round: TDM-staged (ping-pong LDS) WMMA bf16 score GEMM -> causal mask
// (diagonal tile only) -> per-half-row top-8 on all 32 lanes -> sorted merge
// -> neighbor mean -> gelu/mix/momentum update. One wave32 per 16 query rows.
// ---------------------------------------------------------------------------
__global__ __launch_bounds__(32)
void DGN3_round(const float* __restrict__ hprev,          // (B,T,D) f32
                const unsigned short* __restrict__ hbf,   // (B,T,D) bf16 bits of hprev
                float* __restrict__ hnext,                // (B,T,D) f32 out
                unsigned short* __restrict__ hbfnext,     // (B,T,D) bf16 out
                const float* __restrict__ gain,           // (R,D)
                const float* __restrict__ bias,           // (R,D)
                const float* __restrict__ log_mix,        // (R,)
                const float* __restrict__ log_momentum,   // (1,)
                const float* __restrict__ log_scale,      // (1,)
                const float* __restrict__ x,              // (B,T,D) original input
                float* __restrict__ out,                  // (B,T,D) final output
                int r, int is_final) {
    __shared__ __align__(64) unsigned short stage[2][16 * D_]; // 2 x 8KB key tiles
    __shared__ float sc[16 * 17];     // 16x16 score tile, padded stride
    __shared__ float mtv[32 * 8];     // per-lane top-8 values (for final merge)
    __shared__ int   mti[32 * 8];     // per-lane top-8 indices
    __shared__ int   tidx[16 * 8];    // merged per-row top-8 indices
    __shared__ int   tcnt[16];        // per-row valid count

    const int g  = blockIdx.x;
    const int b  = g / TTILES;
    const int it = g % TTILES;        // query tile index
    const int l  = threadIdx.x;
    const int lh = l >> 4;            // 0 / 1 half-wave
    const int lm = l & 15;

    const unsigned stage_off[2] = {
        (unsigned)(size_t)(void*)&stage[0][0],
        (unsigned)(size_t)(void*)&stage[1][0] };
    const unsigned short* keybase = hbf + (size_t)b * T_ * D_;   // key rows, this batch

    // ---- preload A fragments: 16 query rows x 256 K, contiguous-K layout (same as B) ----
    // element e of lane l holds K = 16*(l>>4) + e, M = l&15
    const unsigned short* qbase = hbf + ((size_t)b * T_ + (size_t)it * 16 + lm) * D_;
    v16bf afrag[8];
#pragma unroll
    for (int kk = 0; kk < 8; ++kk) {
        APairBits bits;
        bits.lo = *(const u16x8*)(qbase + 32 * kk + 16 * lh);
        bits.hi = *(const u16x8*)(qbase + 32 * kk + 16 * lh + 8);
        afrag[kk] = __builtin_bit_cast(v16bf, bits);
    }

    // ---- running top-8: lane l owns row (l&15), tile-columns [lh*8, lh*8+8) ----
    float tv[8]; int ti[8];
#pragma unroll
    for (int k = 0; k < 8; ++k) { tv[k] = NEGV; ti[k] = 0; }

    // kick off DMA of key tile 0
    tdm_load_tile(keybase, stage_off[0]);

    for (int jt = 0; jt <= it; ++jt) {
        // issue next tile's DMA; latency hides under WMMA + top-k below
        if (jt < it) {
            tdm_load_tile(keybase + (size_t)(jt + 1) * 16 * D_,
                          stage_off[(jt + 1) & 1]);
            if (jt + 2 <= it)
                __builtin_prefetch(keybase + (size_t)(jt + 2) * 16 * D_ + lm * D_,
                                   0, 3);
            __builtin_amdgcn_s_wait_tensorcnt(1);   // tile jt landed, jt+1 in flight
        } else {
            __builtin_amdgcn_s_wait_tensorcnt(0);   // last tile: drain
        }

        // B fragments from LDS stage: B[k][n]=h_key[n][k]; lane l: N=l&15,
        // K = 16*(l>>4) + e  -> contiguous 32B at row lm, kbase 32*kk+16*lh
        const unsigned short* sb = &stage[jt & 1][0] + lm * D_ + 16 * lh;
        v16bf bfrag[8];
#pragma unroll
        for (int kk = 0; kk < 8; ++kk) {
            u16x16 raw = *(const u16x16*)(sb + 32 * kk);
            bfrag[kk] = __builtin_bit_cast(v16bf, raw);
        }

        v8f c = {0.f, 0.f, 0.f, 0.f, 0.f, 0.f, 0.f, 0.f};
#pragma unroll
        for (int kk = 0; kk < 8; ++kk) {
            c = __builtin_amdgcn_wmma_f32_16x16x32_bf16(
                    false, afrag[kk], false, bfrag[kk], (short)0, c, false, false);
        }

        // spill tile to LDS (C layout: M = v + 8*lh, N = lm); mask only on diagonal
        if (jt == it) {
            const int gj = jt * 16 + lm;
#pragma unroll
            for (int v = 0; v < 8; ++v) {
                const int M   = v + 8 * lh;
                const int row = it * 16 + M;
                float s = c[v];
                if (gj > row) s = NEGV;
                sc[M * 17 + lm] = s;
            }
        } else {
#pragma unroll
            for (int v = 0; v < 8; ++v)
                sc[(v + 8 * lh) * 17 + lm] = c[v];
        }
        __syncthreads();   // single-wave WG: lowers to LDS waits

        // all 32 lanes: 8 candidates each (half a row), insertion into sorted top-8
        {
            const int cbase = jt * 16 + lh * 8;
            for (int cx = 0; cx < 8; ++cx) {
                float v = sc[lm * 17 + lh * 8 + cx];
                if (v > tv[7]) {
                    tv[7] = v; ti[7] = cbase + cx;
#pragma unroll
                    for (int q = 7; q > 0; --q) {
                        if (tv[q] > tv[q - 1]) {
                            float tf = tv[q]; tv[q] = tv[q - 1]; tv[q - 1] = tf;
                            int   tn = ti[q]; ti[q] = ti[q - 1]; ti[q - 1] = tn;
                        }
                    }
                }
            }
        }
        __syncthreads();
    }

    // ---- merge the two per-row sorted top-8 lists (lanes r and r+16) ----
#pragma unroll
    for (int k = 0; k < 8; ++k) { mtv[l * 8 + k] = tv[k]; mti[l * 8 + k] = ti[k]; }
    __syncthreads();

    if (l < 16) {
        int ia = 0, ib = 0, cnt = 0;
#pragma unroll
        for (int k = 0; k < 8; ++k) {
            float va = mtv[l * 8 + ia];
            float vb = mtv[(l + 16) * 8 + ib];
            float vm; int im;
            if (va >= vb) { vm = va; im = mti[l * 8 + ia];        ia++; }
            else          { vm = vb; im = mti[(l + 16) * 8 + ib]; ib++; }
            tidx[l * 8 + k] = im;
            if (vm > NEGV * 0.5f) cnt++;
        }
        tcnt[l] = cnt > 0 ? cnt : 1;
    }
    __syncthreads();

    // ---- aggregation + pointwise update: lane handles dims [l*8, l*8+8) ----
    const float mix = sigm(log_mix[r]);
    const float mom = sigm(log_momentum[0]);
    const float scl = log1pf(expf(log_scale[0])) + 0.01f;
    const int   d0  = l * 8;
    const float* gp = gain + (size_t)r * D_ + d0;
    const float* bp = bias + (size_t)r * D_ + d0;
    const float4 g0 = *(const float4*)gp, g1 = *(const float4*)(gp + 4);
    const float4 b0 = *(const float4*)bp, b1 = *(const float4*)(bp + 4);
    const float gv[8] = {g0.x, g0.y, g0.z, g0.w, g1.x, g1.y, g1.z, g1.w};
    const float bv[8] = {b0.x, b0.y, b0.z, b0.w, b1.x, b1.y, b1.z, b1.w};

    for (int row = 0; row < 16; ++row) {
        const size_t ibase = ((size_t)b * T_ + (size_t)it * 16 + row) * D_ + d0;
        const int cnt = tcnt[row];
        float acc[8];
#pragma unroll
        for (int e = 0; e < 8; ++e) acc[e] = 0.f;
#pragma unroll
        for (int k = 0; k < 8; ++k) {
            if (k < cnt) {
                const int j = tidx[row * 8 + k];
                const float* hp = hprev + ((size_t)b * T_ + j) * D_ + d0;
                float4 p0 = *(const float4*)hp;
                float4 p1 = *(const float4*)(hp + 4);
                acc[0] += p0.x; acc[1] += p0.y; acc[2] += p0.z; acc[3] += p0.w;
                acc[4] += p1.x; acc[5] += p1.y; acc[6] += p1.z; acc[7] += p1.w;
            }
        }
        const float inv = 1.0f / (float)cnt;
        float4 h0 = *(const float4*)(hprev + ibase);
        float4 h1 = *(const float4*)(hprev + ibase + 4);
        const float hv[8] = {h0.x, h0.y, h0.z, h0.w, h1.x, h1.y, h1.z, h1.w};

        float hn[8];
        u16x8 ob;
#pragma unroll
        for (int e = 0; e < 8; ++e) {
            float msg = acc[e] * inv;
            float bl  = mix * hv[e] + (1.0f - mix) * msg;
            float pre = bl * gv[e] + bv[e];
            float ge  = 0.5f * pre * (1.0f + erff(pre * 0.7071067811865475f));
            hn[e]     = mom * hv[e] + (1.0f - mom) * ge;
            ob[e]     = f2bf(hn[e]);
        }
        float4 o0 = {hn[0], hn[1], hn[2], hn[3]};
        float4 o1 = {hn[4], hn[5], hn[6], hn[7]};
        *(float4*)(hnext + ibase)     = o0;
        *(float4*)(hnext + ibase + 4) = o1;
        *(u16x8*)(hbfnext + ibase)    = ob;

        if (is_final) {
            float4 x0 = *(const float4*)(x + ibase);
            float4 x1 = *(const float4*)(x + ibase + 4);
            float4 f0 = {(hn[0] - x0.x) * scl, (hn[1] - x0.y) * scl,
                         (hn[2] - x0.z) * scl, (hn[3] - x0.w) * scl};
            float4 f1 = {(hn[4] - x1.x) * scl, (hn[5] - x1.y) * scl,
                         (hn[6] - x1.z) * scl, (hn[7] - x1.w) * scl};
            *(float4*)(out + ibase)     = f0;
            *(float4*)(out + ibase + 4) = f1;
        }
    }
}

// ---------------------------------------------------------------------------
extern "C" void kernel_launch(void* const* d_in, const int* in_sizes, int n_in,
                              void* d_out, int out_size, void* d_ws, size_t ws_size,
                              hipStream_t stream) {
    (void)in_sizes; (void)n_in; (void)out_size; (void)ws_size;
    const float* x      = (const float*)d_in[0];
    const float* gain   = (const float*)d_in[1];
    const float* bias   = (const float*)d_in[2];
    const float* lmix   = (const float*)d_in[3];
    const float* lmom   = (const float*)d_in[4];
    const float* lscale = (const float*)d_in[5];
    float*       out    = (float*)d_out;

    const size_t N = (size_t)B_ * T_ * D_;     // 2,097,152 elements
    float*          h1  = (float*)d_ws;
    float*          h2  = h1 + N;
    unsigned short* bfA = (unsigned short*)(h2 + N);
    unsigned short* bfB = bfA + N;
    // ws: 2*N*4 + 2*N*2 = 24 MB

    const int ntiles = B_ * TTILES;            // 512 wave-tiles

    DGN3_cvt_bf16<<<(int)((N + 255) / 256), 256, 0, stream>>>(x, bfA, (int)N);

    // round 0: x -> h1
    DGN3_round<<<ntiles, 32, 0, stream>>>(x,  bfA, h1, bfB, gain, bias,
                                          lmix, lmom, lscale, x, out, 0, 0);
    // round 1: h1 -> h2
    DGN3_round<<<ntiles, 32, 0, stream>>>(h1, bfB, h2, bfA, gain, bias,
                                          lmix, lmom, lscale, x, out, 1, 0);
    // round 2: h2 -> h1, fused (h - x) * scale epilogue into d_out
    DGN3_round<<<ntiles, 32, 0, stream>>>(h2, bfA, h1, bfB, gain, bias,
                                          lmix, lmom, lscale, x, out, 2, 1);
}
